// myModel_27341761806472
// MI455X (gfx1250) — matrix operand order repaired and checked by reading, not compiled, not run on hardware
//
#include <hip/hip_runtime.h>
#include <hip/hip_bf16.h>

// ---------------------------------------------------------------------------
// GNN message passing on MI455X (gfx1250), wave32 + WMMA bf16.
//
//   T=4 x [ gather -> selu(concat@Wmsg) -> relu(msg@Wgcn) -> scatter-add ]
//   then per-graph segment sum + small MLP readout.
//
// Edge GEMMs run on v_wmma_f32_16x16x32_bf16 (f32 accumulate), everything
// else in fp32. Link state (25.6MB) is L2-resident; scatter uses f32 atomics.
// ---------------------------------------------------------------------------

typedef float  v8f   __attribute__((ext_vector_type(8)));
typedef __bf16 v16bf __attribute__((ext_vector_type(16)));
typedef __bf16 v8bf  __attribute__((ext_vector_type(8)));

#define WAVES_PER_BLOCK 8
#define DIM 64

__device__ __forceinline__ v8f zero8() {
    v8f z;
#pragma unroll
    for (int i = 0; i < 8; ++i) z[i] = 0.0f;
    return z;
}

__device__ __forceinline__ float selu_f(float x) {
    // jax.nn.selu constants
    const float lam      = 1.0507009873554805f;
    const float lam_alp  = 1.7580993408473766f;   // lam * 1.6732632423543772
    return x > 0.0f ? lam * x : lam_alp * (__expf(x) - 1.0f);
}

__device__ __forceinline__ v16bf load_bfrag(const __bf16* p) {
    v8bf lo = *(const v8bf*)p;
    v8bf hi = *(const v8bf*)(p + 8);
    return __builtin_shufflevector(lo, hi, 0,1,2,3,4,5,6,7,8,9,10,11,12,13,14,15);
}

// ---------------------------------------------------------------------------
// Edge message-passing kernel: one wave per 16-edge tile.
//   out[second[e]] += relu( selu( [in[first[e]], in[second[e]]] @ Wmsg + bmsg )
//                           @ Wgcn + bgcn )
// out must be pre-zeroed.
// ---------------------------------------------------------------------------
__global__ __launch_bounds__(256) void edge_pass_kernel(
    const float* __restrict__ in,      // [nLinks, 64]
    const int*   __restrict__ first,   // [nEdges]
    const int*   __restrict__ second,  // [nEdges]
    const float* __restrict__ Wmsg,    // [128, 64] row-major
    const float* __restrict__ bmsg,    // [64]
    const float* __restrict__ Wgcn,    // [64, 64] row-major
    const float* __restrict__ bgcn,    // [64]
    float*       __restrict__ out,     // [nLinks, 64], pre-zeroed
    int nEdges, int nLinks)
{
    // Pre-swizzled B fragments: 24 frags (16 for GEMM1, 8 for GEMM2),
    // per fragment 32 lanes x 16 bf16, lane stride padded to 24 bf16 (48B)
    // to cut LDS bank conflicts on ds_load_b128.
    __shared__ __align__(16) __bf16 wlds[24 * 32 * 24];
    // Per-wave msg tile for C->A relayout between the two GEMMs.
    __shared__ __align__(16) __bf16 msglds[WAVES_PER_BLOCK][16][72]; // 64 + 8 pad

    const int tid = threadIdx.x;

    // ---- stage weights as bf16 WMMA B-fragments -------------------------
    // B-fragment layout (16x16x32 bf16): lane -> N = n*16 + (lane&15),
// element j<8 -> K = kc*32 + (lane>>4)*8 + j, j>=8 -> K = kc*32 + 16 + (lane>>4)*8 + (j-8).
    for (int it = tid; it < 24 * 32; it += 256) {
        const int frag = it >> 5;
        const int lane = it & 31;
        const int h    = lane >> 4;
        int kc, nn;
        const float* W;
        if (frag < 16) { kc = frag >> 2;        nn = frag & 3;        W = Wmsg; }
        else           { kc = (frag - 16) >> 2; nn = (frag - 16) & 3; W = Wgcn; }
        const int N = nn * 16 + (lane & 15);
        __bf16* dst = &wlds[frag * 768 + lane * 24];
#pragma unroll
        for (int j = 0; j < 16; ++j) {
            const int K = kc * 32 + (j < 8 ? h * 8 + j : 16 + h * 8 + (j - 8));
            dst[j] = (__bf16)W[K * 64 + N];
        }
    }
    __syncthreads();

    const int wave = tid >> 5;
    const int lane = tid & 31;
    const int h    = lane >> 4;   // lane half (0: lanes 0-15, 1: lanes 16-31)
    const int m    = lane & 15;   // row within 16-edge tile / column index

    const int tile = blockIdx.x * WAVES_PER_BLOCK + wave;
    const int e0   = tile * 16;
    if (e0 >= nEdges) return;
    int rowsValid = nEdges - e0; if (rowsValid > 16) rowsValid = 16;

    int eMine = e0 + m; if (eMine >= nEdges) eMine = nEdges - 1;
    int idxF = first[eMine];  if ((unsigned)idxF >= (unsigned)nLinks) idxF = 0;
    int idxS = second[eMine]; if ((unsigned)idxS >= (unsigned)nLinks) idxS = 0;

    const float* rowF = in + (size_t)idxF * DIM;
    const float* rowS = in + (size_t)idxS * DIM;

    // ---- GEMM1: [16x128] @ Wmsg[128x64] -> acc[4] (N tiles) -------------
    v8f acc[4];
#pragma unroll
    for (int n = 0; n < 4; ++n) acc[n] = zero8();

#pragma unroll
    for (int kc = 0; kc < 4; ++kc) {
        // concat K range [kc*32, kc*32+32): kc<2 -> main half, else neigh.
        const float* src = (kc < 2) ? rowF : rowS;
        const int lk = (kc & 1) * 32;
        // A layout (16-bit 16x32): elem j<8 -> K = lk + h*8 + j,
        //                          elem j>=8 -> K = lk + 16 + h*8 + (j-8)
        const float4* s0 = (const float4*)(src + lk + h * 8);
        const float4* s1 = (const float4*)(src + lk + 16 + h * 8);
        const float4 p0 = s0[0], p1 = s0[1];
        const float4 p2 = s1[0], p3 = s1[1];
        v16bf a;
        a[0]=(__bf16)p0.x; a[1]=(__bf16)p0.y; a[2]=(__bf16)p0.z; a[3]=(__bf16)p0.w;
        a[4]=(__bf16)p1.x; a[5]=(__bf16)p1.y; a[6]=(__bf16)p1.z; a[7]=(__bf16)p1.w;
        a[8]=(__bf16)p2.x; a[9]=(__bf16)p2.y; a[10]=(__bf16)p2.z; a[11]=(__bf16)p2.w;
        a[12]=(__bf16)p3.x; a[13]=(__bf16)p3.y; a[14]=(__bf16)p3.z; a[15]=(__bf16)p3.w;
#pragma unroll
        for (int n = 0; n < 4; ++n) {
            const v16bf b = load_bfrag(&wlds[(kc * 4 + n) * 768 + lane * 24]);
            acc[n] = __builtin_amdgcn_wmma_f32_16x16x32_bf16(
                false, a, false, b, (short)0, acc[n], false, false);
        }
    }

    // ---- bias + selu, relayout C -> A via per-wave LDS ------------------
    // C layout: VGPR g, lanes 0-15 -> M=g, lanes 16-31 -> M=g+8; N = n*16+m.
#pragma unroll
    for (int n = 0; n < 4; ++n) {
        const float bv = bmsg[n * 16 + m];
#pragma unroll
        for (int g = 0; g < 8; ++g) {
            const float x = selu_f(acc[n][g] + bv);
            msglds[wave][g + 8 * h][n * 16 + m] = (__bf16)x;
        }
    }
    // DS ops are in-order within a wave: store->load RAW through LDS is safe.

    // ---- GEMM2: msg[16x64] @ Wgcn[64x64] -> acc2[4] ---------------------
    v8f acc2[4];
#pragma unroll
    for (int n = 0; n < 4; ++n) acc2[n] = zero8();

#pragma unroll
    for (int kc = 0; kc < 2; ++kc) {
        const __bf16* arow = &msglds[wave][m][kc * 32 + h * 8];
        const v8bf lo = *(const v8bf*)arow;
        const v8bf hi = *(const v8bf*)(arow + 16);
        const v16bf a = __builtin_shufflevector(lo, hi,
            0,1,2,3,4,5,6,7,8,9,10,11,12,13,14,15);
#pragma unroll
        for (int n = 0; n < 4; ++n) {
            const v16bf b = load_bfrag(&wlds[(16 + kc * 4 + n) * 768 + lane * 24]);
            acc2[n] = __builtin_amdgcn_wmma_f32_16x16x32_bf16(
                false, a, false, b, (short)0, acc2[n], false, false);
        }
    }

    // ---- bias + relu + scatter-add by states_second ---------------------
    int rIdx[8];
#pragma unroll
    for (int g = 0; g < 8; ++g) {
        const int rr = g + 8 * h;            // output row M of this VGPR
        if (rr < rowsValid) {
            int ri = second[e0 + rr];
            rIdx[g] = ((unsigned)ri < (unsigned)nLinks) ? ri : -1;
        } else {
            rIdx[g] = -1;
        }
    }
#pragma unroll
    for (int n = 0; n < 4; ++n) {
        const float bv = bgcn[n * 16 + m];
#pragma unroll
        for (int g = 0; g < 8; ++g) {
            if (rIdx[g] >= 0) {
                const float x = fmaxf(acc2[n][g] + bv, 0.0f);
                atomicAdd(out + (size_t)rIdx[g] * DIM + n * 16 + m, x);
            }
        }
    }
}

// ---------------------------------------------------------------------------
// Per-graph segment sum: emb[gid[l]] += state[l]   (emb pre-zeroed)
// ---------------------------------------------------------------------------
__global__ __launch_bounds__(256) void graph_emb_kernel(
    const float* __restrict__ state, const int* __restrict__ gid,
    float* __restrict__ emb, int nLinks)
{
    const int i = blockIdx.x * 256 + threadIdx.x;
    if (i >= nLinks * DIM) return;
    const int l = i >> 6, c = i & 63;
    atomicAdd(emb + gid[l] * DIM + c, state[i]);
}

// ---------------------------------------------------------------------------
// Readout MLP (tiny): selu(emb@W1+b1), selu(@W2+b2), @W3+b3
// ---------------------------------------------------------------------------
__global__ __launch_bounds__(256) void readout1_kernel(
    const float* __restrict__ emb, const float* __restrict__ W,
    const float* __restrict__ b, float* __restrict__ out)
{
    const int g = blockIdx.x, u = threadIdx.x;   // 256 x 256
    const float* er = emb + g * 64;
    float s = 0.0f;
#pragma unroll 8
    for (int k = 0; k < 64; ++k) s += er[k] * W[k * 256 + u];
    out[g * 256 + u] = selu_f(s + b[u]);
}

__global__ __launch_bounds__(256) void readout2_kernel(
    const float* __restrict__ r1, const float* __restrict__ W,
    const float* __restrict__ b, float* __restrict__ out)
{
    const int g = blockIdx.x, u = threadIdx.x;   // 256 x 256
    const float* rr = r1 + g * 256;
    float s = 0.0f;
#pragma unroll 8
    for (int k = 0; k < 256; ++k) s += rr[k] * W[k * 256 + u];
    out[g * 256 + u] = selu_f(s + b[u]);
}

__global__ __launch_bounds__(256) void readout3_kernel(
    const float* __restrict__ r2, const float* __restrict__ W,
    const float* __restrict__ b, float* __restrict__ out)
{
    const int g = threadIdx.x;                    // 256 graphs, 1 block
    const float* rr = r2 + g * 256;
    float s = 0.0f;
#pragma unroll 8
    for (int k = 0; k < 256; ++k) s += rr[k] * W[k];
    out[g] = s + b[0];
}

// ---------------------------------------------------------------------------
extern "C" void kernel_launch(void* const* d_in, const int* in_sizes, int n_in,
                              void* d_out, int out_size, void* d_ws, size_t ws_size,
                              hipStream_t stream)
{
    const float* states_action = (const float*)d_in[0];
    const float* Wmsg = (const float*)d_in[1];
    const float* bmsg = (const float*)d_in[2];
    const float* Wgcn = (const float*)d_in[3];
    const float* bgcn = (const float*)d_in[4];
    const float* Wr1  = (const float*)d_in[5];
    const float* br1  = (const float*)d_in[6];
    const float* Wr2  = (const float*)d_in[7];
    const float* br2  = (const float*)d_in[8];
    const float* Wr3  = (const float*)d_in[9];
    const float* br3  = (const float*)d_in[10];
    const int* gids   = (const int*)d_in[11];
    const int* first  = (const int*)d_in[12];
    const int* second = (const int*)d_in[13];

    const int nLinks = in_sizes[0] / DIM;
    const int nEdges = in_sizes[12];

    const size_t stateBytes = ((size_t)nLinks * DIM * sizeof(float) + 255) & ~(size_t)255;
    const size_t needed = 2 * stateBytes + 256 * 64 * 4 + 2 * (256 * 256 * 4);
    if (ws_size < needed) return;

    char* ws = (char*)d_ws;
    float* bufA = (float*)ws;
    float* bufB = (float*)(ws + stateBytes);
    char* p = ws + 2 * stateBytes;
    float* emb = (float*)p; p += 256 * 64 * 4;
    float* r1  = (float*)p; p += 256 * 256 * 4;
    float* r2  = (float*)p;

    const int tiles  = (nEdges + 15) / 16;
    const int blocks = (tiles + WAVES_PER_BLOCK - 1) / WAVES_PER_BLOCK;

    const float* cur = states_action;
    float* nxt = bufA;
    for (int t = 0; t < 4; ++t) {
        hipMemsetAsync(nxt, 0, (size_t)nLinks * DIM * sizeof(float), stream);
        edge_pass_kernel<<<blocks, 256, 0, stream>>>(
            cur, first, second, Wmsg, bmsg, Wgcn, bgcn, nxt, nEdges, nLinks);
        cur = nxt;
        nxt = (nxt == bufA) ? bufB : bufA;
    }

    hipMemsetAsync(emb, 0, 256 * 64 * 4, stream);
    const int total = nLinks * DIM;
    graph_emb_kernel<<<(total + 255) / 256, 256, 0, stream>>>(cur, gids, emb, nLinks);
    readout1_kernel<<<256, 256, 0, stream>>>(emb, Wr1, br1, r1);
    readout2_kernel<<<256, 256, 0, stream>>>(r1, Wr2, br2, r2);
    readout3_kernel<<<1, 256, 0, stream>>>(r2, Wr3, br3, (float*)d_out);
}
